// AttentionBlock_66503273611398
// MI455X (gfx1250) — compile-verified, matrix operand order checked by reading, not run on hardware
//
#include <hip/hip_runtime.h>
#include <math.h>

#ifndef M_PI
#define M_PI 3.14159265358979323846
#endif

#define N_TOK   2048
#define HIDDEN  2880
#define NH      64
#define NKV     8
#define HD      64
#define QKV_DIM ((NH + 2 * NKV) * HD)   // 5120
#define Q_SZ    (NH * HD)               // 4096
#define KV_SZ   (NKV * HD)              // 512
#define WINDOW  128
#define SM_SCALE 0.125f                 // 1/sqrt(64)
#define RMS_EPS 1e-5f

typedef __bf16 bf16;
typedef __attribute__((ext_vector_type(16))) __bf16 v16bf;
typedef __attribute__((ext_vector_type(8)))  float  v8f;

// ---------------------------------------------------------------------------
// 16-bit A (16x32) / B (32x16) WMMA fragment layout:
//   lanes 0-15 : K = 0..7  and 16..23
//   lanes 16-31: K = 8..15 and 24..31
// -> two contiguous 8-element runs at (lh*8) and (16 + lh*8).
// ---------------------------------------------------------------------------
__device__ inline v16bf frag_from_lds(const bf16* base, int lh) {
  typedef __attribute__((ext_vector_type(8))) __bf16 v8bf;
  union { uint4 u; v8bf h; } a, b;
  a.u = *(const uint4*)(base + lh * 8);
  b.u = *(const uint4*)(base + 16 + lh * 8);
  v16bf r;
#pragma unroll
  for (int i = 0; i < 8; ++i) { r[i] = a.h[i]; r[i + 8] = b.h[i]; }
  return r;
}

__device__ inline v16bf frag_from_f32(const float* base, int lh, int off) {
  const float* p0 = base + off + lh * 8;
  const float* p1 = base + off + 16 + lh * 8;
  union { float4 v; float f[4]; } a0, a1, b0, b1;
  a0.v = *(const float4*)p0;       a1.v = *(const float4*)(p0 + 4);
  b0.v = *(const float4*)p1;       b1.v = *(const float4*)(p1 + 4);
  v16bf r;
#pragma unroll
  for (int i = 0; i < 4; ++i) {
    r[i]      = (bf16)a0.f[i];  r[i + 4]  = (bf16)a1.f[i];
    r[i + 8]  = (bf16)b0.f[i];  r[i + 12] = (bf16)b1.f[i];
  }
  return r;
}

// ---------------------------------------------------------------------------
// 1) RMSNorm -> bf16 activations
// ---------------------------------------------------------------------------
__global__ __launch_bounds__(256) void rmsnorm_kernel(
    const float* __restrict__ x, const float* __restrict__ scale,
    bf16* __restrict__ t) {
  int row = blockIdx.x;
  const float* xr = x + (size_t)row * HIDDEN;
  float ss = 0.f;
  for (int i = threadIdx.x; i < HIDDEN; i += blockDim.x) { float v = xr[i]; ss += v * v; }
  __shared__ float red[8];
#pragma unroll
  for (int m = 16; m; m >>= 1) ss += __shfl_xor(ss, m, 32);
  int wid = threadIdx.x >> 5, lid = threadIdx.x & 31;
  if (lid == 0) red[wid] = ss;
  __syncthreads();
  if (wid == 0) {
    float v = (lid < 8) ? red[lid] : 0.f;
#pragma unroll
    for (int m = 4; m; m >>= 1) v += __shfl_xor(v, m, 32);
    if (lid == 0) red[0] = v;
  }
  __syncthreads();
  float rinv = rsqrtf(red[0] / (float)HIDDEN + RMS_EPS);
  bf16* tr = t + (size_t)row * HIDDEN;
  for (int i = threadIdx.x; i < HIDDEN; i += blockDim.x)
    tr[i] = (bf16)(xr[i] * rinv * scale[i]);
}

// ---------------------------------------------------------------------------
// 2) Tiled bf16 WMMA GEMM:  C[M,N] = A_bf16[M,K] @ B_f32[K,N] + bias (+res)
//    Block = 256 threads (8 waves).  Tile 128(M) x 64(N), K-step 64.
//    A tile staged via CDNA5 async global->LDS DMA (ASYNCcnt).
//    B tile converted f32->bf16 in registers, stored transposed as k-pairs.
// ---------------------------------------------------------------------------
#define GT_M 128
#define GT_N 64
#define GT_K 64

template <int K, int N>
__global__ __launch_bounds__(256) void gemm_bf16_kernel(
    const bf16* __restrict__ A, const float* __restrict__ B,
    const float* __restrict__ bias, const float* __restrict__ residual,
    float* __restrict__ C) {
  __shared__ bf16 ldsA[GT_M * GT_K];    // 16 KB
  __shared__ bf16 ldsBt[GT_N * GT_K];   // 8 KB, transposed: [n][k]

  const int m0 = blockIdx.y * GT_M;
  const int n0 = blockIdx.x * GT_N;
  const int tid = threadIdx.x;
  const int lane = tid & 31, wv = tid >> 5;
  const int lh = lane >> 4, lm = lane & 15;

  // staging maps (fixed per thread)
  const int a_row = tid >> 1;                 // 0..127
  const int a_kc  = (tid & 1) * 32;           // 0 or 32 (x8 bf16 = 64B chunk)
  const uint a_lds = (uint)(uintptr_t)&ldsA[a_row * GT_K + a_kc];
  const int b_n  = tid & 63;                  // column within tile
  const int b_pg = tid >> 6;                  // 0..3 (k-pair group)

  v8f acc[4] = {};

  for (int k0 = 0; k0 < K; k0 += GT_K) {
    // --- A tile: 128x64 bf16 via async global->LDS (64B per thread) ------
    {
      const bf16* ga = A + (size_t)(m0 + a_row) * K + k0 + a_kc;
      asm volatile(
          "global_load_async_to_lds_b128 %0, %1, off\n\t"
          "global_load_async_to_lds_b128 %0, %1, off offset:16\n\t"
          "global_load_async_to_lds_b128 %0, %1, off offset:32\n\t"
          "global_load_async_to_lds_b128 %0, %1, off offset:48"
          :: "v"(a_lds), "v"(ga) : "memory");
    }
    // --- B tile: load 16 f32 into regs (batched), then pack bf16 pairs ---
    {
      const float* gb = B + (size_t)k0 * N + n0 + b_n;
      float f0[8], f1[8];
#pragma unroll
      for (int j = 0; j < 8; ++j) {
        const int p = b_pg + 4 * j;           // k-pair index 0..31
        f0[j] = gb[(size_t)(2 * p) * N];
        f1[j] = gb[(size_t)(2 * p + 1) * N];
      }
#pragma unroll
      for (int j = 0; j < 8; ++j) {
        const int p = b_pg + 4 * j;
        union { bf16 h[2]; uint u; } pk;
        pk.h[0] = (bf16)f0[j]; pk.h[1] = (bf16)f1[j];
        *(uint*)&ldsBt[b_n * GT_K + 2 * p] = pk.u;
      }
      if (k0 + GT_K < K)
        __builtin_prefetch(&B[(size_t)(k0 + GT_K + b_pg) * N + n0 + b_n], 0, 1);
    }
    asm volatile("s_wait_asynccnt 0" ::: "memory");
    __syncthreads();

    // --- compute: 2 K-fragments x 4 N-tiles = 8 WMMAs per wave -----------
#pragma unroll
    for (int kf = 0; kf < 2; ++kf) {
      v16bf af = frag_from_lds(&ldsA[(wv * 16 + lm) * GT_K + kf * 32], lh);
#pragma unroll
      for (int t = 0; t < 4; ++t) {
        v16bf bfg = frag_from_lds(&ldsBt[(t * 16 + lm) * GT_K + kf * 32], lh);
        acc[t] = __builtin_amdgcn_wmma_f32_16x16x32_bf16(
            false, af, false, bfg, (short)0, acc[t], false, false);
      }
    }
    __syncthreads();
  }

  // --- epilogue: C layout VGPR i -> row = i + 8*lh, col = lm --------------
#pragma unroll
  for (int t = 0; t < 4; ++t) {
    const int col = n0 + t * 16 + lm;
    const float bv = bias[col];
#pragma unroll
    for (int i = 0; i < 8; ++i) {
      const int row = m0 + wv * 16 + lh * 8 + i;
      float v = acc[t][i] + bv;
      if (residual) v += residual[(size_t)row * N + col];
      C[(size_t)row * N + col] = v;
    }
  }
}

// ---------------------------------------------------------------------------
// 3) YaRN RoPE tables + in-place apply on q/k of the qkv buffer
// ---------------------------------------------------------------------------
__global__ void rope_table_kernel(float* __restrict__ cs) {
  int idx = blockIdx.x * blockDim.x + threadIdx.x;
  if (idx >= N_TOK * 32) return;
  int tpos = idx >> 5, d = idx & 31;
  const float base = 150000.0f;
  float freq = powf(base, (float)d / 32.0f);
  float conc = 0.1f * logf(32.0f) + 1.0f;
  float lb = logf(base);
  float lo = 32.0f * logf(4096.0f / (32.0f * 2.0f * (float)M_PI)) / lb;
  float hi = 32.0f * logf(4096.0f / (2.0f * (float)M_PI)) / lb;
  float ramp = fminf(fmaxf(((float)d - lo) / (hi - lo), 0.0f), 1.0f);
  float mfac = 1.0f - ramp;
  float inv_freq = (1.0f / (32.0f * freq)) * (1.0f - mfac) + (1.0f / freq) * mfac;
  float ang = (float)tpos * inv_freq;
  cs[(size_t)tpos * 64 + d]      = cosf(ang) * conc;
  cs[(size_t)tpos * 64 + 32 + d] = sinf(ang) * conc;
}

__global__ void rope_apply_kernel(float* __restrict__ qkv,
                                  const float* __restrict__ cs) {
  const int HEADS = NH + NKV;   // q heads then k heads (contiguous columns)
  int idx = blockIdx.x * blockDim.x + threadIdx.x;
  if (idx >= N_TOK * HEADS * 32) return;
  int d = idx & 31;
  int h = (idx >> 5) % HEADS;
  int tpos = idx / (32 * HEADS);
  float c = cs[(size_t)tpos * 64 + d];
  float s = cs[(size_t)tpos * 64 + 32 + d];
  size_t off = (size_t)tpos * QKV_DIM + h * HD;
  float x1 = qkv[off + d], x2 = qkv[off + d + 32];
  qkv[off + d]      = x1 * c - x2 * s;
  qkv[off + d + 32] = x2 * c + x1 * s;
}

// ---------------------------------------------------------------------------
// 4) Flash attention, one wave per (query head, 16-query tile).
//    Sink column handled exactly: m0 = sink, l0 = 1.
// ---------------------------------------------------------------------------
__global__ __launch_bounds__(128) void attn_kernel(
    const float* __restrict__ qkv, const float* __restrict__ sink,
    bf16* __restrict__ attn) {
  __shared__ bf16 ldsP[4][16 * 32];   // per-wave P scratch (C->A transpose)
  int wv = threadIdx.x >> 5, lane = threadIdx.x & 31;
  int gw = blockIdx.x * 4 + wv;
  int hq = gw >> 7;                    // 128 q-tiles per head
  int qt = gw & 127;
  int q0 = qt * 16;
  int hkv = hq >> 3;                   // Q_MULT = 8
  int lh = lane >> 4, lm = lane & 15;

  // Q fragments (A-layout): rows = queries q0+lm, K = head dim
  const float* qrow = qkv + (size_t)(q0 + lm) * QKV_DIM + hq * HD;
  v16bf aq0 = frag_from_f32(qrow, lh, 0);
  v16bf aq1 = frag_from_f32(qrow, lh, 32);

  float m_i[8], l_i[8];
  float snk = sink[hq];
#pragma unroll
  for (int i = 0; i < 8; ++i) { m_i[i] = snk; l_i[i] = 1.0f; }
  v8f acc[4] = {};

  int lo = q0 - WINDOW; if (lo < 0) lo = 0;
  int j0 = lo & ~31;
  bf16* Pw = ldsP[wv];

  for (int j = j0; j <= q0 + 15; j += 32) {
    // --- scores for two 16-key tiles (K=64 -> 2 chained WMMAs each) ------
    v8f s[2] = {};
#pragma unroll
    for (int tt = 0; tt < 2; ++tt) {
      int key = j + tt * 16 + lm; if (key > N_TOK - 1) key = N_TOK - 1;
      const float* krow = qkv + (size_t)key * QKV_DIM + Q_SZ + hkv * HD;
      v16bf bk0 = frag_from_f32(krow, lh, 0);
      v16bf bk1 = frag_from_f32(krow, lh, 32);
      s[tt] = __builtin_amdgcn_wmma_f32_16x16x32_bf16(
          false, aq0, false, bk0, (short)0, s[tt], false, false);
      s[tt] = __builtin_amdgcn_wmma_f32_16x16x32_bf16(
          false, aq1, false, bk1, (short)0, s[tt], false, false);
    }

    // --- mask + online softmax (rows: i + 8*lh, col: lm) -----------------
    float p0a[8], p1a[8];
#pragma unroll
    for (int i = 0; i < 8; ++i) {
      int row = q0 + lh * 8 + i;
      float sv0, sv1;
      {
        int kp = j + lm;
        float sc = s[0][i] * SM_SCALE;
        sv0 = (kp <= row && row - kp <= WINDOW) ? sc : -INFINITY;
      }
      {
        int kp = j + 16 + lm;
        float sc = s[1][i] * SM_SCALE;
        sv1 = (kp <= row && row - kp <= WINDOW) ? sc : -INFINITY;
      }
      float v = fmaxf(sv0, sv1);
#pragma unroll
      for (int msk = 1; msk < 16; msk <<= 1) v = fmaxf(v, __shfl_xor(v, msk, 32));
      float mnew = fmaxf(m_i[i], v);
      float alpha = expf(m_i[i] - mnew);
      m_i[i] = mnew;
      float p0 = expf(sv0 - mnew);
      float p1 = expf(sv1 - mnew);
      float rs = p0 + p1;
#pragma unroll
      for (int msk = 1; msk < 16; msk <<= 1) rs += __shfl_xor(rs, msk, 32);
      l_i[i] = l_i[i] * alpha + rs;
#pragma unroll
      for (int t = 0; t < 4; ++t) acc[t][i] *= alpha;
      p0a[i] = p0; p1a[i] = p1;
    }

    // --- P: C-layout -> A-layout via per-wave LDS ------------------------
#pragma unroll
    for (int i = 0; i < 8; ++i) {
      int prow = lh * 8 + i;
      Pw[prow * 32 + lm]      = (bf16)p0a[i];
      Pw[prow * 32 + 16 + lm] = (bf16)p1a[i];
    }
    asm volatile("s_wait_dscnt 0" ::: "memory");   // CDNA5 split DS counter
    v16bf pf = frag_from_lds(&Pw[lm * 32], lh);

    // --- PV: 32 keys x 64 dims (4 N-tiles); batch strided V loads --------
#pragma unroll
    for (int t = 0; t < 4; ++t) {
      int dcol = t * 16 + lm;
      const float* vbase = qkv + Q_SZ + KV_SZ + hkv * HD + dcol;
      float fv[16];
#pragma unroll
      for (int e = 0; e < 8; ++e) {
        int k1 = j + lh * 8 + e;        if (k1 > N_TOK - 1) k1 = N_TOK - 1;
        int k2 = j + 16 + lh * 8 + e;   if (k2 > N_TOK - 1) k2 = N_TOK - 1;
        fv[e]     = vbase[(size_t)k1 * QKV_DIM];
        fv[e + 8] = vbase[(size_t)k2 * QKV_DIM];
      }
      v16bf vf;
#pragma unroll
      for (int e = 0; e < 16; ++e) vf[e] = (bf16)fv[e];
      acc[t] = __builtin_amdgcn_wmma_f32_16x16x32_bf16(
          false, pf, false, vf, (short)0, acc[t], false, false);
    }
  }

  // --- finalize: divide by denom (includes sink term), write bf16 --------
#pragma unroll
  for (int t = 0; t < 4; ++t) {
#pragma unroll
    for (int i = 0; i < 8; ++i) {
      int row = q0 + lh * 8 + i;
      int d = t * 16 + lm;
      attn[(size_t)row * Q_SZ + hq * HD + d] = (bf16)(acc[t][i] / l_i[i]);
    }
  }
}

// ---------------------------------------------------------------------------
// Launch: rmsnorm -> QKV GEMM -> RoPE -> attention -> O-proj GEMM (+residual)
// Workspace: 11.8MB t_bf16 + 40MB qkv f32 + 0.5MB rope + 16.8MB attn ~ 70MB
// ---------------------------------------------------------------------------
extern "C" void kernel_launch(void* const* d_in, const int* in_sizes, int n_in,
                              void* d_out, int out_size, void* d_ws, size_t ws_size,
                              hipStream_t stream) {
  const float* x     = (const float*)d_in[0];
  const float* scale = (const float*)d_in[1];
  const float* sink  = (const float*)d_in[2];
  const float* w_qkv = (const float*)d_in[3];
  const float* b_qkv = (const float*)d_in[4];
  const float* w_o   = (const float*)d_in[5];
  const float* b_o   = (const float*)d_in[6];
  float* out = (float*)d_out;

  char* ws = (char*)d_ws;
  bf16*  t_bf   = (bf16*)ws;   ws += (size_t)N_TOK * HIDDEN * sizeof(bf16);
  float* qkv    = (float*)ws;  ws += (size_t)N_TOK * QKV_DIM * sizeof(float);
  float* ropecs = (float*)ws;  ws += (size_t)N_TOK * 64 * sizeof(float);
  bf16*  attn   = (bf16*)ws;

  rmsnorm_kernel<<<N_TOK, 256, 0, stream>>>(x, scale, t_bf);
  rope_table_kernel<<<(N_TOK * 32 + 255) / 256, 256, 0, stream>>>(ropecs);

  dim3 g1(QKV_DIM / GT_N, N_TOK / GT_M);
  gemm_bf16_kernel<HIDDEN, QKV_DIM><<<g1, 256, 0, stream>>>(
      t_bf, w_qkv, b_qkv, nullptr, qkv);

  int nrope = N_TOK * (NH + NKV) * 32;
  rope_apply_kernel<<<(nrope + 255) / 256, 256, 0, stream>>>(qkv, ropecs);

  attn_kernel<<<(NH * 128) / 4, 128, 0, stream>>>(qkv, sink, attn);

  dim3 g2(HIDDEN / GT_N, N_TOK / GT_M);
  gemm_bf16_kernel<Q_SZ, HIDDEN><<<g2, 256, 0, stream>>>(
      attn, w_o, b_o, x, out);
}
